// SelfAttention_56581899158204
// MI455X (gfx1250) — matrix operand order repaired and checked by reading, not compiled, not run on hardware
//
#include <hip/hip_runtime.h>
#include <cstdint>

// ---------------------------------------------------------------------------
// Self-attention forward on MI455X (gfx1250, wave32, WMMA f32_16x16x32_f16)
//   B=2, S=2048, D=1024, H=16, d=64
// Pipeline: fp32->f16 convert; QKV GEMM (16x64 tiles/wave); cooperative
// flash attention with async global->LDS double-buffered K/V staging.
// ---------------------------------------------------------------------------

typedef __attribute__((ext_vector_type(16))) _Float16 v16h;
typedef __attribute__((ext_vector_type(8)))  _Float16 v8h;
typedef __attribute__((ext_vector_type(4)))  _Float16 v4h;
typedef __attribute__((ext_vector_type(8)))  float    v8f;

#define D_MODEL 1024
#define SEQ     2048
#define NH      16
#define DH      64
#define BATCH   2
#define BH      (BATCH * NH)   // 32 (b,h) pairs
#define KB      64             // key block for attention

// Two b128 loads -> one 16-half WMMA operand.
static __device__ __forceinline__ v16h ld2x8(const _Float16* p0, const _Float16* p1) {
    v8h lo = *reinterpret_cast<const v8h*>(p0);
    v8h hi = *reinterpret_cast<const v8h*>(p1);
    return __builtin_shufflevector(lo, hi, 0,1,2,3,4,5,6,7,8,9,10,11,12,13,14,15);
}

static __device__ __forceinline__ v8f wmma16(v16h a, v16h b, v8f c) {
    return __builtin_amdgcn_wmma_f32_16x16x32_f16(false, a, false, b, (short)0, c,
                                                  false, false);
}

// Per-lane async copy of 16B global -> 16B LDS (ASYNCcnt-tracked, ISA §15.18.3).
static __device__ __forceinline__ void async_g2l_b128(unsigned lds_byte_off,
                                                      const void* gaddr) {
    asm volatile("global_load_async_to_lds_b128 %0, %1, off"
                 :: "v"(lds_byte_off), "v"(gaddr) : "memory");
}

// ---------------------------------------------------------------------------
// fp32 -> f16 conversion (sizes are multiples of 4)
// ---------------------------------------------------------------------------
__global__ void cvt_f32_to_f16(const float* __restrict__ src,
                               _Float16* __restrict__ dst, int n) {
    int i = (blockIdx.x * blockDim.x + threadIdx.x) * 4;
    if (i < n) {
        const float4 f = *reinterpret_cast<const float4*>(src + i);
        v4h h;
        h[0] = (_Float16)f.x; h[1] = (_Float16)f.y;
        h[2] = (_Float16)f.z; h[3] = (_Float16)f.w;
        *reinterpret_cast<v4h*>(dst + i) = h;
    }
}

// ---------------------------------------------------------------------------
// QKV projection: y = x @ W^T.  One wave per 16(M) x 64(N) tile == one head
// slice.  A operand (x rows) reused across 4 N-subtiles -> 4 WMMAs per A load.
// Outputs: Q (x 1/8) and K as [bh][s][d]; V transposed [bh][d][s].
// ---------------------------------------------------------------------------
__global__ __launch_bounds__(128) void qkv_wmma(
    const _Float16* __restrict__ xh,    // [B*S][D]
    const _Float16* __restrict__ wh,    // [3][D][D] (q,k,v)
    _Float16* __restrict__ qh,          // [BH][SEQ][DH]
    _Float16* __restrict__ kh,          // [BH][SEQ][DH]
    _Float16* __restrict__ vth)         // [BH][DH][SEQ]
{
    const int lane = threadIdx.x & 31;
    const int r = lane & 15, g = lane >> 4;

    int idx = blockIdx.x * 4 + (threadIdx.x >> 5);
    const int TM = (BATCH * SEQ) / 16;   // 256 m-tiles
    const int TN = D_MODEL / 64;         // 16 n-blocks (one head each)
    int tm = idx % TM; idx /= TM;
    int tn = idx % TN; idx /= TN;
    const int mat = idx;                 // 0=Q 1=K 2=V
    const int m0 = tm * 16, n0 = tn * 64;

    const _Float16* A  = xh + (size_t)(m0 + r) * D_MODEL;
    const _Float16* Bw = wh + (size_t)mat * D_MODEL * D_MODEL
                            + (size_t)(n0 + r) * D_MODEL + g * 16;

    v8f acc[4] = {};
    #pragma unroll 2
    for (int kk = 0; kk < D_MODEL; kk += 32) {
        const v16h a = ld2x8(A + kk + g * 16, A + kk + g * 16 + 8);
        #pragma unroll
        for (int nt = 0; nt < 4; ++nt) {
            const _Float16* bp = Bw + (size_t)nt * 16 * D_MODEL + kk;
            v16h b = ld2x8(bp, bp + 8);
            acc[nt] = wmma16(a, b, acc[nt]);
        }
    }

    // D tile: VGPR v -> row (v + 8*g), col r; subtile nt -> d = nt*16 + r
    const int bb = m0 >> 11;             // batch (S==2048)
    const int s0 = m0 & (SEQ - 1);
    const int hh = n0 >> 6;              // one head per n-block
    const int bh = bb * NH + hh;

    if (mat == 2) {
        #pragma unroll
        for (int nt = 0; nt < 4; ++nt) {
            v8h pk;
            #pragma unroll
            for (int v = 0; v < 8; ++v) pk[v] = (_Float16)acc[nt][v];
            *reinterpret_cast<v8h*>(
                vth + ((size_t)bh * DH + nt * 16 + r) * SEQ + s0 + 8 * g) = pk;
        }
    } else {
        _Float16* outp = (mat == 0) ? qh : kh;
        const float sc = (mat == 0) ? 0.125f : 1.0f;   // fold 1/sqrt(DH) into Q
        #pragma unroll
        for (int nt = 0; nt < 4; ++nt)
            #pragma unroll
            for (int v = 0; v < 8; ++v)
                outp[((size_t)bh * SEQ + (s0 + v + 8 * g)) * DH + nt * 16 + r] =
                    (_Float16)(acc[nt][v] * sc);
    }
}

// ---------------------------------------------------------------------------
// Cooperative causal flash attention.
//   Block = 4 waves = 4 consecutive 16-query tiles of one (b,h).
//   K/V 64-key blocks staged global->LDS with async b128 copies, double
//   buffered (ASYNCcnt); every wave processes every key block (last block is
//   its diagonal block), so barriers are divergence-free.
//   Per key block per wave: 8 WMMAs for S=QK^T, fp32 online softmax
//   (shfl_xor butterfly within 16-lane halves), P bounced via LDS to
//   A-layout, 8 WMMAs for O += P V.
// ---------------------------------------------------------------------------
__global__ __launch_bounds__(128) void flash_attn_wmma(
    const _Float16* __restrict__ qh,
    const _Float16* __restrict__ kh,
    const _Float16* __restrict__ vth,
    float* __restrict__ out)
{
    __shared__ alignas(128) _Float16 ldsK[2][KB][DH];    // 2 x 8 KB
    __shared__ alignas(128) _Float16 ldsV[2][DH][KB];    // 2 x 8 KB
    __shared__ alignas(64)  _Float16 lds_p[4][16][KB];   // 4 x 2 KB

    const int tid  = threadIdx.x;
    const int lane = tid & 31;
    const int w    = tid >> 5;
    const int r = lane & 15, g = lane >> 4;

    const int bh      = blockIdx.x >> 5;          // SEQ/KB = 32 groups per head
    const int base_m0 = (blockIdx.x & 31) * 64;   // 64 queries per block
    const int m0      = base_m0 + 16 * w;         // this wave's query tile

    const _Float16* Kb = kh  + (size_t)bh * SEQ * DH;
    const _Float16* Vb = vth + (size_t)bh * DH * SEQ;

    // Cooperative async stage of one 64-key K block (64x64 halfs) and the
    // matching V block (64 d-rows x 64 keys). 512 16B chunks each, 128
    // threads x 4 chunks -> 8 async instructions per wave per stage.
    auto stage = [&](int buf, int j0) {
        const unsigned kbase = (unsigned)(uintptr_t)&ldsK[buf][0][0];
        const unsigned vbase = (unsigned)(uintptr_t)&ldsV[buf][0][0];
        #pragma unroll
        for (int i = 0; i < 4; ++i) {
            const int c   = tid + 128 * i;   // 0..511
            const int row = c >> 3;          // 0..63
            const int col = (c & 7) * 8;     // half index, 16B granules
            async_g2l_b128(kbase + (unsigned)(row * DH + col) * 2,
                           Kb + (size_t)(j0 + row) * DH + col);
            async_g2l_b128(vbase + (unsigned)(row * KB + col) * 2,
                           Vb + (size_t)row * SEQ + j0 + col);
        }
    };

    // Q A-operands for both 32-wide K-steps over d=64, loaded once.
    const _Float16* Qp = qh + ((size_t)bh * SEQ + m0 + r) * DH;
    const v16h aq0 = ld2x8(Qp + g * 16,      Qp + g * 16 + 8);
    const v16h aq1 = ld2x8(Qp + 32 + g * 16, Qp + 40 + g * 16);

    v8f o0 = {}, o1 = {}, o2 = {}, o3 = {};
    float mrun[8], lrun[8];
    #pragma unroll
    for (int v = 0; v < 8; ++v) { mrun[v] = -1e30f; lrun[v] = 0.0f; }

    const int nblocks = base_m0 / KB + 1;   // uniform across the block
    stage(0, 0);

    for (int jb = 0; jb < nblocks; ++jb) {
        const int j0  = jb * KB;
        const int cur = jb & 1;
        if (jb + 1 < nblocks) {
            stage(cur ^ 1, j0 + KB);                        // prefetch next
            asm volatile("s_wait_asynccnt 0x8" ::: "memory"); // retire cur batch
        } else {
            asm volatile("s_wait_asynccnt 0x0" ::: "memory");
        }
        __syncthreads();   // all waves' staged portions visible

        // ---- scores: 4 key subtiles x 2 K-steps = 8 WMMAs ----
        v8f sc[4];
        #pragma unroll
        for (int t = 0; t < 4; ++t) {
            const _Float16* kp = &ldsK[cur][t * 16 + r][g * 16];
            v16h b0 = ld2x8(kp,      kp + 8);
            v16h b1 = ld2x8(kp + 32, kp + 40);
            v8f s = {};
            s = wmma16(aq0, b0, s);
            s = wmma16(aq1, b1, s);
            sc[t] = s;
        }

        // ---- causal mask (diagonal block only) + online softmax ----
        const bool needmask = (jb == nblocks - 1);
        #pragma unroll
        for (int v = 0; v < 8; ++v) {
            float a0 = sc[0][v], a1 = sc[1][v], a2 = sc[2][v], a3 = sc[3][v];
            if (needmask) {
                const int q = m0 + v + 8 * g;
                if (j0 + r      > q) a0 = -1e30f;
                if (j0 + 16 + r > q) a1 = -1e30f;
                if (j0 + 32 + r > q) a2 = -1e30f;
                if (j0 + 48 + r > q) a3 = -1e30f;
            }
            float t = fmaxf(fmaxf(a0, a1), fmaxf(a2, a3));
            t = fmaxf(t, __shfl_xor(t, 1, 32));
            t = fmaxf(t, __shfl_xor(t, 2, 32));
            t = fmaxf(t, __shfl_xor(t, 4, 32));
            t = fmaxf(t, __shfl_xor(t, 8, 32));
            const float mnew  = fmaxf(mrun[v], t);
            const float scale = __expf(mrun[v] - mnew);
            const float p0 = __expf(a0 - mnew);
            const float p1 = __expf(a1 - mnew);
            const float p2 = __expf(a2 - mnew);
            const float p3 = __expf(a3 - mnew);
            float rs = (p0 + p1) + (p2 + p3);
            rs += __shfl_xor(rs, 1, 32);
            rs += __shfl_xor(rs, 2, 32);
            rs += __shfl_xor(rs, 4, 32);
            rs += __shfl_xor(rs, 8, 32);
            lrun[v] = lrun[v] * scale + rs;
            mrun[v] = mnew;
            o0[v] *= scale; o1[v] *= scale; o2[v] *= scale; o3[v] *= scale;
            const int pr = v + 8 * g;
            lds_p[w][pr][r]      = (_Float16)p0;
            lds_p[w][pr][16 + r] = (_Float16)p1;
            lds_p[w][pr][32 + r] = (_Float16)p2;
            lds_p[w][pr][48 + r] = (_Float16)p3;
        }

        // D-layout -> A-layout bounce (per-wave LDS; same-wave DS is in-order)
        __builtin_amdgcn_wave_barrier();
        const _Float16* pp = &lds_p[w][r][0];
        const v16h ap0 = ld2x8(pp + g * 16,      pp + g * 16 + 8);
        const v16h ap1 = ld2x8(pp + 32 + g * 16, pp + 40 + g * 16);
        __builtin_amdgcn_wave_barrier();

        // ---- O += P @ V : 4 d-tiles x 2 key K-steps = 8 WMMAs ----
        {
            const _Float16* v0 = &ldsV[cur][r][g * 16];
            v16h b;
            b = ld2x8(v0,      v0 + 8);  o0 = wmma16(ap0, b, o0);
            b = ld2x8(v0 + 32, v0 + 40); o0 = wmma16(ap1, b, o0);
            const _Float16* v1 = v0 + 16 * KB;
            b = ld2x8(v1,      v1 + 8);  o1 = wmma16(ap0, b, o1);
            b = ld2x8(v1 + 32, v1 + 40); o1 = wmma16(ap1, b, o1);
            const _Float16* v2 = v0 + 32 * KB;
            b = ld2x8(v2,      v2 + 8);  o2 = wmma16(ap0, b, o2);
            b = ld2x8(v2 + 32, v2 + 40); o2 = wmma16(ap1, b, o2);
            const _Float16* v3 = v0 + 48 * KB;
            b = ld2x8(v3,      v3 + 8);  o3 = wmma16(ap0, b, o3);
            b = ld2x8(v3 + 32, v3 + 40); o3 = wmma16(ap1, b, o3);
        }
        __syncthreads();   // reads done before buffer reuse two iterations on
    }

    // ---- epilogue: normalize, scatter to [B][S][H*DH] fp32 ----
    const int bb = bh >> 4, hh = bh & 15;
    #pragma unroll
    for (int v = 0; v < 8; ++v) {
        const int q = m0 + v + 8 * g;
        const float inv = 1.0f / lrun[v];
        float* op = out + ((size_t)bb * SEQ + q) * D_MODEL + hh * DH + r;
        op[0]  = o0[v] * inv;
        op[16] = o1[v] * inv;
        op[32] = o2[v] * inv;
        op[48] = o3[v] * inv;
    }
}

// ---------------------------------------------------------------------------
// Host launch
// ---------------------------------------------------------------------------
extern "C" void kernel_launch(void* const* d_in, const int* in_sizes, int n_in,
                              void* d_out, int out_size, void* d_ws, size_t ws_size,
                              hipStream_t stream) {
    (void)in_sizes; (void)n_in; (void)out_size; (void)ws_size;
    const float* x  = (const float*)d_in[0];
    const float* Wq = (const float*)d_in[1];
    const float* Wk = (const float*)d_in[2];
    const float* Wv = (const float*)d_in[3];
    float* out = (float*)d_out;

    // workspace layout (f16), ~40 MB total
    _Float16* xh  = (_Float16*)d_ws;                            // [4096][1024]
    _Float16* wh  = xh + (size_t)BATCH * SEQ * D_MODEL;         // [3][1024][1024]
    _Float16* qh  = wh + (size_t)3 * D_MODEL * D_MODEL;         // [BH][SEQ][DH]
    _Float16* kh  = qh + (size_t)BH * SEQ * DH;                 // [BH][SEQ][DH]
    _Float16* vth = kh + (size_t)BH * SEQ * DH;                 // [BH][DH][SEQ]

    {
        int n = BATCH * SEQ * D_MODEL;
        cvt_f32_to_f16<<<(n / 4 + 255) / 256, 256, 0, stream>>>(x, xh, n);
    }
    {
        int n = D_MODEL * D_MODEL;
        cvt_f32_to_f16<<<(n / 4 + 255) / 256, 256, 0, stream>>>(Wq, wh, n);
        cvt_f32_to_f16<<<(n / 4 + 255) / 256, 256, 0, stream>>>(
            Wk, wh + (size_t)D_MODEL * D_MODEL, n);
        cvt_f32_to_f16<<<(n / 4 + 255) / 256, 256, 0, stream>>>(
            Wv, wh + (size_t)2 * D_MODEL * D_MODEL, n);
    }
    {
        int tiles = 3 * (D_MODEL / 64) * ((BATCH * SEQ) / 16);  // 12288 waves
        qkv_wmma<<<tiles / 4, 128, 0, stream>>>(xh, wh, qh, kh, vth);
    }
    {
        int blocks = BH * (SEQ / 64);                           // 1024 blocks
        flash_attn_wmma<<<blocks, 128, 0, stream>>>(qh, kh, vth, out);
    }
}
